// NeuralMemory_38225208934923
// MI455X (gfx1250) — compile-verified, matrix operand order checked by reading, not run on hardware
//
#include <hip/hip_runtime.h>
#include <hip/hip_bf16.h>

#define D_MODEL 1024
#define D_MEM   64
#define CAP     65536
#define NROWS   2048          // B*S
#define THETA   0.01f
#define SCALE   0.125f        // 1/sqrt(D_MEM)
#define GRADSC  (2.0f / (float)(NROWS * D_MEM))

typedef __attribute__((ext_vector_type(16))) __bf16 v16bf;
typedef __attribute__((ext_vector_type(8)))  float  v8f;

// ---------- bf16 helpers (native converts; v_cvt_pk_bf16_f32 on gfx1250) ----------
__device__ __forceinline__ __bf16 f2bf(float f) { return (__bf16)f; }
__device__ __forceinline__ void split_bf(float f, __bf16 &hi, __bf16 &lo) {
  float fh = __uint_as_float(__float_as_uint(f) & 0xFFFF0000u);  // exact bf16 value
  hi = (__bf16)fh;                                               // exact convert
  lo = (__bf16)(f - fh);                                         // residual, RNE
}
__device__ __forceinline__ v8f wmma_bf(v16bf a, v16bf b, v8f c) {
  return __builtin_amdgcn_wmma_f32_16x16x32_bf16(false, a, false, b, (short)0, c, false, false);
}
struct BFPair { v16bf hi, lo; };

// ---------- fragment loaders (ISA 7.12.2 layouts; bf16 16x16x32) ----------
// A (16M x 32K) from f32 row-major: lane row = lane%16; elem e ->
//   K = k_base + (e/8)*16 + (lane/16)*8 + (e%8)
// Per-lane data = two contiguous 8-float runs -> four b128 loads.
__device__ __forceinline__ BFPair load_a_hl(const float* src, int ld, int row_base, int k_base) {
  int lane = threadIdx.x & 31;
  const float* rp = src + (size_t)(row_base + (lane & 15)) * ld + k_base + ((lane >> 4) << 3);
  float4 f0 = *(const float4*)(rp);
  float4 f1 = *(const float4*)(rp + 4);
  float4 f2 = *(const float4*)(rp + 16);
  float4 f3 = *(const float4*)(rp + 20);
  float f[16] = {f0.x, f0.y, f0.z, f0.w, f1.x, f1.y, f1.z, f1.w,
                 f2.x, f2.y, f2.z, f2.w, f3.x, f3.y, f3.z, f3.w};
  BFPair r;
#pragma unroll
  for (int e = 0; e < 16; ++e) {
    __bf16 h2, l2; split_bf(f[e], h2, l2);
    r.hi[e] = h2; r.lo[e] = l2;
  }
  return r;
}
// B = mat^T slice (logical B[k][n] = mat[n][k]) : same gather as A with n<->row
__device__ __forceinline__ BFPair load_bT_hl(const float* mat, int ld, int n_base, int k_base) {
  return load_a_hl(mat, ld, n_base, k_base);
}
// B (32K x 16N) from f32 row-major mat[k][n] (stride-ld gather; inherently scalar)
__device__ __forceinline__ v16bf load_b(const float* mat, int ld, int k_base, int n_base) {
  int lane = threadIdx.x & 31;
  int n = n_base + (lane & 15);
  int hb = (lane >> 4) << 3;
  v16bf b;
#pragma unroll
  for (int e = 0; e < 16; ++e) {
    int k = k_base + ((e >> 3) << 4) + hb + (e & 7);
    b[e] = f2bf(mat[(size_t)k * ld + n]);
  }
  return b;
}
// A from a [16][32] bf16 LDS tile (two 16B runs -> two ds_load_b128)
struct U32x8 { uint4 a, b; };
__device__ __forceinline__ v16bf load_a_bf_lds(const __bf16* src) {
  int lane = threadIdx.x & 31;
  const __bf16* rp = src + (lane & 15) * 32 + ((lane >> 4) << 3);
  U32x8 u;
  u.a = *(const uint4*)(rp);
  u.b = *(const uint4*)(rp + 16);
  return __builtin_bit_cast(v16bf, u);
}
// A^T from a [32][16] bf16 LDS tile (A[m][k] = src[k][m])
__device__ __forceinline__ v16bf load_aT_bf_lds(const __bf16* src) {
  int lane = threadIdx.x & 31;
  int m2 = lane & 15;
  int hb = (lane >> 4) << 3;
  v16bf a;
#pragma unroll
  for (int e = 0; e < 16; ++e) {
    int k = ((e >> 3) << 4) + hb + (e & 7);
    a[e] = src[k * 16 + m2];
  }
  return a;
}

// ---------- cross-lane reductions ----------
__device__ __forceinline__ float half16_max(float v) {
#pragma unroll
  for (int m2 = 1; m2 <= 8; m2 <<= 1) v = fmaxf(v, __shfl_xor(v, m2, 32));
  return v;
}
__device__ __forceinline__ float half16_sum(float v) {
#pragma unroll
  for (int m2 = 1; m2 <= 8; m2 <<= 1) v += __shfl_xor(v, m2, 32);
  return v;
}
__device__ __forceinline__ float wave_sum(float v) {
#pragma unroll
  for (int m2 = 1; m2 <= 16; m2 <<= 1) v += __shfl_xor(v, m2, 32);
  return v;
}
__device__ __forceinline__ void lds_wave_fence() {
  __builtin_amdgcn_wave_barrier();
  asm volatile("s_wait_dscnt 0" ::: "memory");
  __builtin_amdgcn_wave_barrier();
}

// ================= K1: k = x@Wk+bk, v = x@Wv+bv (hi/lo bf16 WMMA) =================
__global__ __launch_bounds__(32) void k_proj(const float* __restrict__ x,
    const float* __restrict__ Wk, const float* __restrict__ bk,
    const float* __restrict__ Wv, const float* __restrict__ bv,
    float* __restrict__ kbuf, float* __restrict__ vbuf) {
  int row_base = blockIdx.x * 16;
  int t = blockIdx.y;                                // 0..3 -> k cols, 4..7 -> v cols
  const float* W    = (t < 4) ? Wk : Wv;
  const float* bias = (t < 4) ? bk : bv;
  float* out        = (t < 4) ? kbuf : vbuf;
  int ncol = (t & 3) * 16;
  v8f acc = {};
  for (int kc = 0; kc < D_MODEL; kc += 32) {
    BFPair a = load_a_hl(x, D_MODEL, row_base, kc);
    int lane = threadIdx.x & 31;
    int n = ncol + (lane & 15);
    int hb = (lane >> 4) << 3;
    v16bf bh, bl;
#pragma unroll
    for (int e = 0; e < 16; ++e) {
      int k = kc + ((e >> 3) << 4) + hb + (e & 7);
      __bf16 h2, l2; split_bf(W[(size_t)k * D_MEM + n], h2, l2);
      bh[e] = h2; bl[e] = l2;
    }
    acc = wmma_bf(a.lo, bh, acc);
    acc = wmma_bf(a.hi, bl, acc);
    acc = wmma_bf(a.hi, bh, acc);
  }
  int lane = threadIdx.x & 31;
  int col = ncol + (lane & 15);
  int h = lane >> 4;
  float bb = bias[col];
#pragma unroll
  for (int j = 0; j < 8; ++j)
    out[(size_t)(row_base + j + 8 * h) * D_MEM + col] = acc[j] + bb;
}

// ================= K2: flash forward: m, Z, retrieved, G_r, rowdot, loss =================
// 128 blocks x 16 rows; 8 waves/block, each wave owns a C stripe of 8192 slots.
__global__ __launch_bounds__(256) void k_forward(
    const float* __restrict__ mem, const float* __restrict__ kbuf,
    const float* __restrict__ vbuf, float* __restrict__ grbuf,
    float* __restrict__ mrow, float* __restrict__ zrow,
    float* __restrict__ rowdot, float* __restrict__ scal) {
  __shared__ __align__(16) __bf16 p_lds[8][16][32];
  __shared__ float  lds_r[8][16][64];
  __shared__ float  lds_m[8][16], lds_z[8][16];

  int wave = threadIdx.x >> 5;
  int lane = threadIdx.x & 31;
  int h = lane >> 4;
  int row_base = blockIdx.x * 16;

  BFPair ak0 = load_a_hl(kbuf, D_MEM, row_base, 0);
  BFPair ak1 = load_a_hl(kbuf, D_MEM, row_base, 32);

  v8f r0 = {}, r1 = {}, r2 = {}, r3 = {};
  float m[8], z[8];
#pragma unroll
  for (int j = 0; j < 8; ++j) { m[j] = -3.0e38f; z[j] = 0.f; }

  int c_start = wave * (CAP / 8);
  for (int it = 0; it < (CAP / 8) / 32; ++it) {
    int c0 = c_start + it * 32;
    // scores 16x32 = k(16x64) . M_chunk^T (hi/lo split)
    v8f s0 = {}, s1 = {};
    {
      BFPair b0 = load_bT_hl(mem, D_MEM, c0, 0);
      BFPair b1 = load_bT_hl(mem, D_MEM, c0, 32);
      s0 = wmma_bf(ak0.lo, b0.hi, s0); s0 = wmma_bf(ak0.hi, b0.lo, s0); s0 = wmma_bf(ak0.hi, b0.hi, s0);
      s0 = wmma_bf(ak1.lo, b1.hi, s0); s0 = wmma_bf(ak1.hi, b1.lo, s0); s0 = wmma_bf(ak1.hi, b1.hi, s0);
      b0 = load_bT_hl(mem, D_MEM, c0 + 16, 0);
      b1 = load_bT_hl(mem, D_MEM, c0 + 16, 32);
      s1 = wmma_bf(ak0.lo, b0.hi, s1); s1 = wmma_bf(ak0.hi, b0.lo, s1); s1 = wmma_bf(ak0.hi, b0.hi, s1);
      s1 = wmma_bf(ak1.lo, b1.hi, s1); s1 = wmma_bf(ak1.hi, b1.lo, s1); s1 = wmma_bf(ak1.hi, b1.hi, s1);
    }
    // online softmax per row (rows j+8h; stats replicated over 16-lane half)
#pragma unroll
    for (int j = 0; j < 8; ++j) {
      float a = s0[j] * SCALE, b = s1[j] * SCALE;
      float mx = half16_max(fmaxf(a, b));
      float nm = fmaxf(m[j], mx);
      float corr = __expf(m[j] - nm);
      float p0 = __expf(a - nm), p1 = __expf(b - nm);
      z[j] = z[j] * corr + half16_sum(p0 + p1);
      m[j] = nm;
      r0[j] *= corr; r1[j] *= corr; r2[j] *= corr; r3[j] *= corr;
      int rr = h * 8 + j;
      p_lds[wave][rr][lane & 15]        = f2bf(p0);
      p_lds[wave][rr][16 + (lane & 15)] = f2bf(p1);
    }
    lds_wave_fence();
    v16bf ap = load_a_bf_lds(&p_lds[wave][0][0]);       // P as A (D->A transpose via LDS)
    __builtin_amdgcn_wave_barrier();
    r0 = wmma_bf(ap, load_b(mem, D_MEM, c0, 0),  r0);   // r += P . M_chunk
    r1 = wmma_bf(ap, load_b(mem, D_MEM, c0, 16), r1);
    r2 = wmma_bf(ap, load_b(mem, D_MEM, c0, 32), r2);
    r3 = wmma_bf(ap, load_b(mem, D_MEM, c0, 48), r3);
  }

  // merge 8 wave-partials
  if ((lane & 15) == 0) {
#pragma unroll
    for (int j = 0; j < 8; ++j) { lds_m[wave][h * 8 + j] = m[j]; lds_z[wave][h * 8 + j] = z[j]; }
  }
#pragma unroll
  for (int j = 0; j < 8; ++j) {
    int rr = h * 8 + j;
    lds_r[wave][rr][lane & 15]        = r0[j];
    lds_r[wave][rr][16 + (lane & 15)] = r1[j];
    lds_r[wave][rr][32 + (lane & 15)] = r2[j];
    lds_r[wave][rr][48 + (lane & 15)] = r3[j];
  }
  __syncthreads();

  int rt = threadIdx.x >> 4;                 // row 0..15, 16 threads (a half-wave) each
  int gr = row_base + rt;
  float mf = -3.0e38f;
#pragma unroll
  for (int w2 = 0; w2 < 8; ++w2) mf = fmaxf(mf, lds_m[w2][rt]);
  float ew[8]; float zf = 0.f;
#pragma unroll
  for (int w2 = 0; w2 < 8; ++w2) { ew[w2] = __expf(lds_m[w2][rt] - mf); zf += lds_z[w2][rt] * ew[w2]; }
  float inv = 1.f / zf;
  float lpart = 0.f, dpart = 0.f;
#pragma unroll
  for (int cc = 0; cc < 4; ++cc) {
    int col = (threadIdx.x & 15) * 4 + cc;
    float rs = 0.f;
#pragma unroll
    for (int w2 = 0; w2 < 8; ++w2) rs += lds_r[w2][rt][col] * ew[w2];
    float retr = rs * inv;
    float diff = retr - vbuf[(size_t)gr * D_MEM + col];
    float g = diff * GRADSC;
    grbuf[(size_t)gr * D_MEM + col] = g;
    lpart += diff * diff;
    dpart += g * retr;                        // rowdot_n = G_r . retrieved  (analytic)
  }
  lpart = half16_sum(lpart); dpart = half16_sum(dpart);
  if ((threadIdx.x & 15) == 0) {
    mrow[gr] = mf; zrow[gr] = zf; rowdot[gr] = dpart;
    atomicAdd(&scal[0], lpart);
  }
}

// ================= K3: backward, slot-parallel: grad_M = w^T G_r + (1/8) Gs^T k ===========
// 512 blocks x 128 slots (8 waves x 16); loop over N in LDS-staged chunks of 32 rows.
__global__ __launch_bounds__(256) void k_backward(
    const float* __restrict__ mem, const float* __restrict__ kbuf,
    const float* __restrict__ grbuf, const float* __restrict__ mrow,
    const float* __restrict__ zrow, const float* __restrict__ rowdot,
    float* __restrict__ grad, float* __restrict__ scal) {
  __shared__ __align__(16) float  k_ch[32][D_MEM];
  __shared__ __align__(16) float  g_ch[32][D_MEM];
  __shared__ float  st_m[32], st_iz[32], st_rd[32];
  __shared__ __align__(16) __bf16 wt[8][32][16];
  __shared__ __align__(16) __bf16 ct[8][32][16];

  int wave = threadIdx.x >> 5, lane = threadIdx.x & 31, h = lane >> 4;
  int c_base = blockIdx.x * 128 + wave * 16;

  BFPair bm0 = load_bT_hl(mem, D_MEM, c_base, 0);   // M_tile^T (K = d), resident in regs
  BFPair bm1 = load_bT_hl(mem, D_MEM, c_base, 32);

  v8f acc[4] = {{}, {}, {}, {}};

  for (int n0 = 0; n0 < NROWS; n0 += 32) {
    for (int i = threadIdx.x * 4; i < 32 * D_MEM; i += 256 * 4) {
      *(float4*)&k_ch[i >> 6][i & 63] = *(const float4*)&kbuf[(size_t)(n0 << 6) + i];
      *(float4*)&g_ch[i >> 6][i & 63] = *(const float4*)&grbuf[(size_t)(n0 << 6) + i];
    }
    if (threadIdx.x < 32) {
      st_m[threadIdx.x]  = mrow[n0 + threadIdx.x];
      st_iz[threadIdx.x] = 1.f / zrow[n0 + threadIdx.x];
      st_rd[threadIdx.x] = rowdot[n0 + threadIdx.x];
    }
    __syncthreads();

#pragma unroll
    for (int ms = 0; ms < 2; ++ms) {
      BFPair ak  = load_a_hl(&k_ch[ms * 16][0], D_MEM, 0, 0);
      BFPair ak2 = load_a_hl(&k_ch[ms * 16][0], D_MEM, 0, 32);
      BFPair ag  = load_a_hl(&g_ch[ms * 16][0], D_MEM, 0, 0);
      BFPair ag2 = load_a_hl(&g_ch[ms * 16][0], D_MEM, 0, 32);
      v8f s = {}, g = {};
      s = wmma_bf(ak.lo, bm0.hi, s);  s = wmma_bf(ak.hi, bm0.lo, s);  s = wmma_bf(ak.hi, bm0.hi, s);
      s = wmma_bf(ak2.lo, bm1.hi, s); s = wmma_bf(ak2.hi, bm1.lo, s); s = wmma_bf(ak2.hi, bm1.hi, s);
      g = wmma_bf(ag.lo, bm0.hi, g);  g = wmma_bf(ag.hi, bm0.lo, g);  g = wmma_bf(ag.hi, bm0.hi, g);
      g = wmma_bf(ag2.lo, bm1.hi, g); g = wmma_bf(ag2.hi, bm1.lo, g); g = wmma_bf(ag2.hi, bm1.hi, g);
#pragma unroll
      for (int j = 0; j < 8; ++j) {
        int nl = ms * 16 + j + 8 * h;
        float w = __expf(s[j] * SCALE - st_m[nl]) * st_iz[nl];
        float coef = w * (g[j] - st_rd[nl]) * SCALE;   // Gs/8 element
        wt[wave][nl][lane & 15] = f2bf(w);
        ct[wave][nl][lane & 15] = f2bf(coef);
      }
    }
    lds_wave_fence();
    v16bf aw = load_aT_bf_lds(&wt[wave][0][0]);        // w^T   (16 slots x 32 n)
    v16bf ac = load_aT_bf_lds(&ct[wave][0][0]);        // Gs^T/8
#pragma unroll
    for (int nd = 0; nd < 4; ++nd) {
      acc[nd] = wmma_bf(aw, load_b(&g_ch[0][0], D_MEM, 0, nd * 16), acc[nd]);  // + w^T G_r
      acc[nd] = wmma_bf(ac, load_b(&k_ch[0][0], D_MEM, 0, nd * 16), acc[nd]);  // + Gs^T k /8
    }
    __syncthreads();
  }

  float ss = 0.f;
#pragma unroll
  for (int nd = 0; nd < 4; ++nd)
#pragma unroll
    for (int j = 0; j < 8; ++j) {
      int c = c_base + j + 8 * h;
      int d = nd * 16 + (lane & 15);
      float v = acc[nd][j];
      grad[(size_t)c * D_MEM + d] = v;
      ss += v * v;
    }
  ss = wave_sum(ss);
  if (lane == 0) atomicAdd(&scal[1], ss);
}

// ================= K4: gates =================
__global__ __launch_bounds__(256) void k_pool(const float* __restrict__ x, float* __restrict__ pooled) {
  int t = blockIdx.x * 256 + threadIdx.x;             // 0..2047
  const float* p = x + (size_t)(t >> 10) * D_MODEL * 1024 + (t & 1023);
  float s = 0.f;
  for (int i = 0; i < 1024; ++i) s += p[(size_t)i * D_MODEL];
  pooled[t] = s * (1.0f / 1024.0f);
}

__global__ __launch_bounds__(128) void k_gates(const float* __restrict__ pooled,
    const float* fw1, const float* fb1, const float* fw2, const float* fb2,
    const float* dw1, const float* db1, const float* dw2, const float* db2,
    float* __restrict__ scal) {
  __shared__ float gv[4];
  int t = threadIdx.x;
  int gsel = t >> 6, b = (t >> 5) & 1, gh = t & 31;   // each (gsel,b) is one full wave
  const float* w1 = gsel ? dw1 : fw1; const float* b1 = gsel ? db1 : fb1;
  const float* w2 = gsel ? dw2 : fw2; const float* b2 = gsel ? db2 : fb2;
  const float* pp = pooled + b * D_MODEL;
  float a = b1[gh];
  for (int i = 0; i < D_MODEL; ++i) a += pp[i] * w1[i * 32 + gh];
  float hsil = a / (1.f + __expf(-a));                 // SiLU
  float c = wave_sum(hsil * w2[gh]);
  if (gh == 0) gv[(gsel << 1) | b] = 1.f / (1.f + __expf(-(c + b2[0])));
  __syncthreads();
  if (t == 0) { scal[2] = 0.5f * (gv[0] + gv[1]); scal[3] = 0.5f * (gv[2] + gv[3]); }
}

// ================= K5: clip + momentum + write M_new, S_new, loss =================
__global__ __launch_bounds__(256) void k_final(const float* __restrict__ mem,
    const float* __restrict__ momS, const float* __restrict__ grad,
    const float* __restrict__ scal, float* __restrict__ out) {
  size_t i = (size_t)blockIdx.x * 256 + threadIdx.x;
  float clip = fminf(1.f, 1.f / sqrtf(scal[1]));
  float alpha = scal[2], eta = scal[3];
  float g = grad[i] * clip;
  float s_new = eta * momS[i] - THETA * g;
  float m_new = (1.f - alpha) * mem[i] + s_new;
  out[1 + i] = m_new;
  out[1 + (size_t)CAP * D_MEM + i] = s_new;           // overwrites staged grad in-place
  if (i == 0) out[0] = scal[0] * (1.0f / (float)(NROWS * D_MEM));
}

extern "C" void kernel_launch(void* const* d_in, const int* in_sizes, int n_in,
                              void* d_out, int out_size, void* d_ws, size_t ws_size,
                              hipStream_t stream) {
  const float* x    = (const float*)d_in[0];
  const float* mem  = (const float*)d_in[1];
  const float* momS = (const float*)d_in[2];
  const float* Wk   = (const float*)d_in[3];
  const float* bk   = (const float*)d_in[4];
  const float* Wv   = (const float*)d_in[5];
  const float* bv   = (const float*)d_in[6];
  const float* fw1  = (const float*)d_in[7];
  const float* fb1  = (const float*)d_in[8];
  const float* fw2  = (const float*)d_in[9];
  const float* fb2  = (const float*)d_in[10];
  const float* dw1  = (const float*)d_in[11];
  const float* db1  = (const float*)d_in[12];
  const float* dw2  = (const float*)d_in[13];
  const float* db2  = (const float*)d_in[14];
  float* out = (float*)d_out;
  float* ws  = (float*)d_ws;

  float* kbuf   = ws;                 // 2048*64
  float* vbuf   = kbuf  + NROWS * D_MEM;
  float* grbuf  = vbuf  + NROWS * D_MEM;
  float* mrowp  = grbuf + NROWS * D_MEM;     // 2048
  float* zrowp  = mrowp + NROWS;
  float* rdp    = zrowp + NROWS;
  float* pooled = rdp   + NROWS;             // 2048
  float* scal   = pooled + NROWS;            // [loss, norm2, alpha, eta]
  float* gradb  = out + 1 + (size_t)CAP * D_MEM;   // stage grad in S_new slot

  hipMemsetAsync(scal, 0, 4 * sizeof(float), stream);
  k_proj    <<<dim3(NROWS / 16, 8), 32,  0, stream>>>(x, Wk, bk, Wv, bv, kbuf, vbuf);
  k_pool    <<<NROWS / 256, 256,        0, stream>>>(x, pooled);
  k_gates   <<<1, 128,                  0, stream>>>(pooled, fw1, fb1, fw2, fb2, dw1, db1, dw2, db2, scal);
  k_forward <<<NROWS / 16, 256,         0, stream>>>(mem, kbuf, vbuf, grbuf, mrowp, zrowp, rdp, scal);
  k_backward<<<CAP / 128, 256,          0, stream>>>(mem, kbuf, grbuf, mrowp, zrowp, rdp, gradb, scal);
  k_final   <<<(CAP * D_MEM) / 256, 256, 0, stream>>>(mem, momS, gradb, scal, out);
}
